// GCNNet_81312320848104
// MI455X (gfx1250) — compile-verified, matrix operand order checked by reading, not run on hardware
//
#include <hip/hip_runtime.h>
#include <hip/hip_bf16.h>
#include <math.h>

// ---------------------------------------------------------------------------
// 2-layer GCN on MI455X (gfx1250, wave32, WMMA).
//   layer: out = D^-1/2 (A+I) D^-1/2 (X W) + b
// Pipeline:
//   k_init     : deg=1 (self loop), zero accumulators
//   k_deg      : atomic degree count over edges
//   k_dinv     : deg -> rsqrt(deg) in place
//   k_gemm1    : xw = bf16wmma(X[100000,512] @ W1[512,16]) f32 accum
//   k_agg1     : edge scatter-add of xw[src]*dinv[src]*dinv[dst] (lane=feature)
//   k_combine1 : h = relu(acc1 + dinv^2*xw + b1)  (self loop folded in)
//   k_gemm2    : hw = bf16wmma(h[100000,16] @ W2[16,40]) K,N zero-padded
//   k_agg2     : edge scatter-add, 40 feats, 8 lanes/edge
//   k_final    : logits = acc2 + dinv^2*hw + b2 ; log_softmax(40)
// ---------------------------------------------------------------------------

typedef __attribute__((ext_vector_type(16))) __bf16 v16bf;
typedef __attribute__((ext_vector_type(8)))  float  v8f;

#define NN     100000
#define F_IN   512
#define H_DIM  16
#define C_DIM  40
#define NTILES (NN / 16)   // 6250, exact

// 16-bit A/B fragment K pattern (ISA 7.12.2): lane khalf = lane>>4
__device__ __forceinline__ int kpat(int khalf, int e) {
    return (e < 8) ? (khalf * 8 + e) : (16 + khalf * 8 + (e - 8));
}

// ---------------------------------------------------------------------- init
__global__ void k_init(float* __restrict__ deg, float* __restrict__ acc1,
                       float* __restrict__ acc2) {
    int tid = blockIdx.x * blockDim.x + threadIdx.x;
    if (tid < NN)          deg[tid]  = 1.0f;   // self loop
    if (tid < NN * H_DIM)  acc1[tid] = 0.0f;
    if (tid < NN * C_DIM)  acc2[tid] = 0.0f;
}

__global__ void k_deg(const int* __restrict__ dst, float* __restrict__ deg, int E) {
    int e = blockIdx.x * blockDim.x + threadIdx.x;
    if (e >= E) return;
    atomicAdd(&deg[dst[e]], 1.0f);
}

__global__ void k_dinv(float* __restrict__ deg) {
    int n = blockIdx.x * blockDim.x + threadIdx.x;
    if (n >= NN) return;
    deg[n] = rsqrtf(deg[n]);   // deg >= 1 always (self loop)
}

// --------------------------------------------------------------- GEMM1: X@W1
// one wave per 16x16 output tile; K=512 in 16 steps of 32; W1 staged in LDS
__global__ __launch_bounds__(256) void k_gemm1(const float* __restrict__ x,
                                               const float* __restrict__ W1,
                                               float* __restrict__ xw) {
    __shared__ __bf16 sW[F_IN * H_DIM];  // 16 KB, shared by 8 waves
    for (int i = threadIdx.x; i < F_IN * H_DIM; i += 256)
        sW[i] = (__bf16)W1[i];
    __syncthreads();

    int tile = blockIdx.x * 8 + (threadIdx.x >> 5);
    if (tile >= NTILES) return;
    int lane  = threadIdx.x & 31;
    int m     = lane & 15;        // A row / C col index
    int khalf = lane >> 4;
    const float* __restrict__ xrow = x + (size_t)(tile * 16 + m) * F_IN;

    v8f c = {};
    for (int kb = 0; kb < F_IN / 32; ++kb) {
        v16bf a, b;
#pragma unroll
        for (int e = 0; e < 16; ++e) {
            int k = kb * 32 + kpat(khalf, e);
            a[e] = (__bf16)xrow[k];          // two contiguous 32B runs/lane
            b[e] = sW[k * H_DIM + m];        // ds_load from staged W1
        }
        c = __builtin_amdgcn_wmma_f32_16x16x32_bf16(
                false, a, false, b, (short)0, c, false, false);
    }
#pragma unroll
    for (int j = 0; j < 8; ++j) {
        int mm = j + 8 * khalf;              // C layout: VGPR j, lane half
        xw[(size_t)(tile * 16 + mm) * H_DIM + m] = c[j];
    }
}

// ----------------------------------------------------------- edge agg, 16 f
__global__ void k_agg1(const int* __restrict__ src, const int* __restrict__ dst,
                       const float* __restrict__ dinv, const float* __restrict__ xw,
                       float* __restrict__ acc1, int E) {
    int tid = blockIdx.x * blockDim.x + threadIdx.x;
    if (tid >= E * 16) return;
    int e = tid >> 4, f = tid & 15;          // lane == feature -> coalesced
    int s = src[e], d = dst[e];
    float w = dinv[s] * dinv[d];
    atomicAdd(&acc1[(size_t)d * H_DIM + f], xw[(size_t)s * H_DIM + f] * w);
}

__global__ void k_combine1(const float* __restrict__ xw, const float* __restrict__ dinv,
                           const float* __restrict__ b1, float* __restrict__ acc1) {
    int tid = blockIdx.x * blockDim.x + threadIdx.x;
    if (tid >= NN * H_DIM) return;
    int n = tid >> 4, f = tid & 15;
    float di = dinv[n];
    float v  = acc1[tid] + di * di * xw[tid] + b1[f];
    acc1[tid] = fmaxf(v, 0.0f);              // acc1 becomes h
}

// --------------------------------------------------------------- GEMM2: h@W2
// K=16 padded to 32 with zeros; N=40 over 3 tiles of 16 (cols>=40 masked)
__global__ __launch_bounds__(256) void k_gemm2(const float* __restrict__ h,
                                               const float* __restrict__ W2,
                                               float* __restrict__ hw) {
    int tile = blockIdx.x * 8 + (threadIdx.x >> 5);
    if (tile >= NTILES) return;
    int lane  = threadIdx.x & 31;
    int m     = lane & 15;
    int khalf = lane >> 4;
    const float* __restrict__ hrow = h + (size_t)(tile * 16 + m) * H_DIM;

    v16bf a;
#pragma unroll
    for (int e = 0; e < 16; ++e)             // e>=8 -> K in [16,32): zero pad
        a[e] = (e < 8) ? (__bf16)hrow[khalf * 8 + e] : (__bf16)0.0f;

    for (int t = 0; t < 3; ++t) {
        int col = t * 16 + m;
        v16bf b;
#pragma unroll
        for (int e = 0; e < 16; ++e) {
            int k = khalf * 8 + e;           // valid only for e<8
            b[e] = (e < 8 && col < C_DIM) ? (__bf16)W2[k * C_DIM + col]
                                          : (__bf16)0.0f;
        }
        v8f c = {};
        c = __builtin_amdgcn_wmma_f32_16x16x32_bf16(
                false, a, false, b, (short)0, c, false, false);
#pragma unroll
        for (int j = 0; j < 8; ++j) {
            int mm = j + 8 * khalf;
            if (col < C_DIM)
                hw[(size_t)(tile * 16 + mm) * C_DIM + col] = c[j];
        }
    }
}

// ----------------------------------------------------------- edge agg, 40 f
__global__ void k_agg2(const int* __restrict__ src, const int* __restrict__ dst,
                       const float* __restrict__ dinv, const float* __restrict__ hw,
                       float* __restrict__ acc2, int E) {
    int tid = blockIdx.x * blockDim.x + threadIdx.x;
    if (tid >= E * 8) return;
    int e = tid >> 3, f0 = tid & 7;          // 8 lanes/edge, 5 feats each
    int s = src[e], d = dst[e];
    float w = dinv[s] * dinv[d];
#pragma unroll
    for (int j = 0; j < 5; ++j) {
        int f = f0 + 8 * j;
        atomicAdd(&acc2[(size_t)d * C_DIM + f], hw[(size_t)s * C_DIM + f] * w);
    }
}

// --------------------------------------------- bias + self loop + logsoftmax
__global__ void k_final(const float* __restrict__ acc2, const float* __restrict__ hw,
                        const float* __restrict__ dinv, const float* __restrict__ b2,
                        float* __restrict__ out) {
    int n = blockIdx.x * blockDim.x + threadIdx.x;
    if (n >= NN) return;
    float di = dinv[n], self = di * di;
    float l[C_DIM];
    float mx = -INFINITY;
#pragma unroll
    for (int f = 0; f < C_DIM; ++f) {
        float v = acc2[(size_t)n * C_DIM + f] + self * hw[(size_t)n * C_DIM + f] + b2[f];
        l[f] = v;
        mx = fmaxf(mx, v);
    }
    float s = 0.0f;
#pragma unroll
    for (int f = 0; f < C_DIM; ++f) s += __expf(l[f] - mx);
    float lse = __logf(s) + mx;
#pragma unroll
    for (int f = 0; f < C_DIM; ++f) out[(size_t)n * C_DIM + f] = l[f] - lse;
}

// ---------------------------------------------------------------------------
extern "C" void kernel_launch(void* const* d_in, const int* in_sizes, int n_in,
                              void* d_out, int out_size, void* d_ws, size_t ws_size,
                              hipStream_t stream) {
    const float* x  = (const float*)d_in[0];
    const int*   ei = (const int*)  d_in[1];      // [2, E]
    const float* W1 = (const float*)d_in[2];      // [512, 16]
    const float* b1 = (const float*)d_in[3];      // [16]
    const float* W2 = (const float*)d_in[4];      // [16, 40]
    const float* b2 = (const float*)d_in[5];      // [40]
    float* out = (float*)d_out;

    const int E = in_sizes[1] / 2;
    const int* src = ei;
    const int* dst = ei + E;

    // workspace layout (floats): deg|xw|acc1(h)|hw|acc2 = 113*N floats
    float* ws   = (float*)d_ws;
    float* deg  = ws;                 // N        (becomes dinv in place)
    float* xw   = ws + (size_t)NN;        // 16N
    float* acc1 = ws + (size_t)17 * NN;   // 16N  (becomes h)
    float* hw   = ws + (size_t)33 * NN;   // 40N
    float* acc2 = ws + (size_t)73 * NN;   // 40N
    (void)ws_size; (void)n_in; (void)out_size;

    const int B = 256;
    // init + normalization (shared by both layers)
    k_init<<<(NN * C_DIM + B - 1) / B, B, 0, stream>>>(deg, acc1, acc2);
    k_deg <<<(E + B - 1) / B, B, 0, stream>>>(dst, deg, E);
    k_dinv<<<(NN + B - 1) / B, B, 0, stream>>>(deg);

    // layer 1
    k_gemm1<<<(NTILES + 7) / 8, B, 0, stream>>>(x, W1, xw);
    k_agg1 <<<((size_t)E * 16 + B - 1) / B, B, 0, stream>>>(src, dst, deg, xw, acc1, E);
    k_combine1<<<(NN * H_DIM + B - 1) / B, B, 0, stream>>>(xw, deg, b1, acc1);

    // layer 2
    k_gemm2<<<(NTILES + 7) / 8, B, 0, stream>>>(acc1, W2, hw);
    k_agg2 <<<((size_t)E * 8 + B - 1) / B, B, 0, stream>>>(src, dst, deg, hw, acc2, E);
    k_final<<<(NN + B - 1) / B, B, 0, stream>>>(acc2, hw, deg, b2, out);
}
